// SACActor_6451040878721
// MI455X (gfx1250) — compile-verified, matrix operand order checked by reading, not run on hardware
//
#include <hip/hip_runtime.h>

// ---------------------------------------------------------------------------
// SAC actor forward for MI455X (gfx1250, wave32).
//  - persistent sLSTM scan kernel: recurrent weights LDS-resident in f16,
//    gate GEMM via v_wmma_f32_16x16x32_f16 (f32 accumulate)
//  - flash-style fused attention per (batch, head)
//  - small VALU kernels for projections / LayerNorm / MLP heads
// ---------------------------------------------------------------------------

#define BATCH  128
#define SEQ    512
#define HID    128
#define FIN    64
#define INFO_D 13
#define ACT    3

// d_out flat offsets (floats): mu | log_std | hs[2,B,H] | cs | ns
#define MU_OFF 0
#define LS_OFF (BATCH * ACT)                 // 384
#define HS_OFF (2 * BATCH * ACT)             // 768
#define CS_OFF (HS_OFF + 2 * BATCH * HID)    // 33536
#define NS_OFF (HS_OFF + 4 * BATCH * HID)    // 66304

typedef _Float16 v16h __attribute__((ext_vector_type(16)));
typedef float    v8f  __attribute__((ext_vector_type(8)));

__device__ __forceinline__ float wave_sum(float v) {
#pragma unroll
  for (int off = 16; off > 0; off >>= 1) v += __shfl_xor(v, off, 32);
  return v;
}
__device__ __forceinline__ float wave_max(float v) {
#pragma unroll
  for (int off = 16; off > 0; off >>= 1) v = fmaxf(v, __shfl_xor(v, off, 32));
  return v;
}
__device__ __forceinline__ float gelu_exact(float x) {
  return 0.5f * x * (1.0f + erff(x * 0.70710678118654752f));
}

// ---------------------------------------------------------------------------
// K1: cur[b,s,:] = LN(x[b,s,:] @ in_proj_w + in_proj_b)   (one wave per row)
// ---------------------------------------------------------------------------
__global__ __launch_bounds__(256) void inproj_ln_kernel(
    const float* __restrict__ x, const float* __restrict__ w,
    const float* __restrict__ bias, const float* __restrict__ ng,
    const float* __restrict__ nb, float* __restrict__ cur) {
  int row  = blockIdx.x * 8 + (threadIdx.x >> 5);  // 0 .. B*S-1
  int lane = threadIdx.x & 31;
  const float* xr = x + (size_t)row * FIN;
  float val[4];
#pragma unroll
  for (int q = 0; q < 4; ++q) val[q] = bias[lane * 4 + q];
  for (int c = 0; c < FIN; ++c) {
    float xv = xr[c];
    const float* wr = w + c * HID + lane * 4;
#pragma unroll
    for (int q = 0; q < 4; ++q) val[q] += xv * wr[q];
  }
  float m = wave_sum(val[0] + val[1] + val[2] + val[3]) * (1.0f / HID);
  float vv = 0.f;
#pragma unroll
  for (int q = 0; q < 4; ++q) { float d = val[q] - m; vv += d * d; }
  float rstd = rsqrtf(wave_sum(vv) * (1.0f / HID) + 1e-5f);
  float* crow = cur + (size_t)row * HID;
#pragma unroll
  for (int q = 0; q < 4; ++q) {
    int j = lane * 4 + q;
    crow[j] = (val[q] - m) * rstd * ng[j] + nb[j];
  }
}

// ---------------------------------------------------------------------------
// K2: persistent sLSTM layer. Grid = B/16 workgroups, 512 threads (16 waves).
// cur is updated IN PLACE: cur[b,t,:] <- h_t + cur[b,t,:].
// LDS: cwB f16 [512][256] (N-major, WMMA-B-friendly) 256KB
//      A   f16 [16][256]  ([xn_t | h])                 8KB
//      g   f32 [16][512]                               32KB
//      c,n f32 [16][128]                               16KB      => 312KB
// ---------------------------------------------------------------------------
__global__ __launch_bounds__(512) void slstm_kernel(
    float* __restrict__ cur, const float* __restrict__ cw,
    const float* __restrict__ cb, const float* __restrict__ lng,
    const float* __restrict__ lnb, float* __restrict__ out, int layer) {
  extern __shared__ char smem[];
  _Float16* cwB = (_Float16*)smem;                       // [512][256] n-major
  _Float16* A   = cwB + 512 * 256;                       // [16][256]
  float*    g   = (float*)(A + 16 * 256);                // [16][512]
  float*    cS  = g + 16 * 512;                          // [16][128]
  float*    nS  = cS + 16 * 128;                         // [16][128]

  const int tid  = threadIdx.x;
  const int wave = tid >> 5;     // 0..15 == batch row within tile
  const int lane = tid & 31;
  const int b0   = blockIdx.x * 16;
  const int row  = wave;

  // pack cw ([256][512] row-major f32) -> cwB[n*256 + k] f16
  for (int idx = tid; idx < 256 * 512; idx += 512) {
    int k = idx >> 9, n = idx & 511;
    cwB[n * 256 + k] = (_Float16)cw[idx];
  }
  // init c=0, n=1, A h-half = 0
  for (int idx = tid; idx < 16 * HID; idx += 512) {
    cS[idx] = 0.f; nS[idx] = 1.f;
    int r = idx >> 7, j = idx & 127;
    A[r * 256 + 128 + j] = (_Float16)0.f;
  }
  // per-thread constants (feature cols lane*4..lane*4+3)
  float lg[4], lb[4], cbi[4], cbf[4], cbo[4], cbz[4];
#pragma unroll
  for (int q = 0; q < 4; ++q) {
    int j = lane * 4 + q;
    lg[q] = lng[j]; lb[q] = lnb[j];
    cbi[q] = cb[j]; cbf[q] = cb[128 + j];
    cbo[q] = cb[256 + j]; cbz[q] = cb[384 + j];
  }
  __syncthreads();

  for (int t = 0; t < SEQ; ++t) {
    // ---- phase 1: xn = LN(x_t); write f16 into A[:,0:128]
    float xr[4];
    float* xrow = cur + (((size_t)(b0 + row)) * SEQ + t) * HID;
    float s = 0.f;
#pragma unroll
    for (int q = 0; q < 4; ++q) { xr[q] = xrow[lane * 4 + q]; s += xr[q]; }
    float m = wave_sum(s) * (1.0f / HID);
    float vv = 0.f;
#pragma unroll
    for (int q = 0; q < 4; ++q) { float d = xr[q] - m; vv += d * d; }
    float rstd = rsqrtf(wave_sum(vv) * (1.0f / HID) + 1e-5f);
#pragma unroll
    for (int q = 0; q < 4; ++q) {
      float xn = (xr[q] - m) * rstd * lg[q] + lb[q];
      A[row * 256 + lane * 4 + q] = (_Float16)xn;
    }
    __syncthreads();

    // ---- phase 2: g[16,512] = A[16,256] @ cw  via WMMA f16->f32
    {
      const int mrow = lane & 15;
      const int hi   = lane >> 4;   // 0: K 0..7/16..23, 1: +8
#pragma unroll
      for (int tile = 0; tile < 2; ++tile) {
        const int n0   = (wave * 2 + tile) * 16;
        const int ncol = n0 + (lane & 15);
        v8f acc = {};
#pragma unroll
        for (int kk = 0; kk < 256; kk += 32) {
          const _Float16* Ap = A   + mrow * 256 + kk + hi * 8;
          const _Float16* Bp = cwB + (size_t)ncol * 256 + kk + hi * 8;
          v16h a, bm;
#pragma unroll
          for (int vv2 = 0; vv2 < 8; ++vv2) {
            int off = (vv2 < 4) ? (2 * vv2) : (16 + 2 * (vv2 - 4));
            a[2 * vv2]      = Ap[off];
            a[2 * vv2 + 1]  = Ap[off + 1];
            bm[2 * vv2]     = Bp[off];
            bm[2 * vv2 + 1] = Bp[off + 1];
          }
          acc = __builtin_amdgcn_wmma_f32_16x16x32_f16(
              false, a, false, bm, (short)0, acc, false, false);
        }
        const int mb = hi * 8;
#pragma unroll
        for (int r = 0; r < 8; ++r)
          g[(mb + r) * 512 + n0 + (lane & 15)] = acc[r];
      }
    }
    __syncthreads();

    // ---- phase 3: gates + state update + in-place residual
    const bool last = (t == SEQ - 1);
#pragma unroll
    for (int q = 0; q < 4; ++q) {
      int j = lane * 4 + q;
      float gi = g[row * 512 + j]       + cbi[q];
      float gf = g[row * 512 + 128 + j] + cbf[q];
      float go = g[row * 512 + 256 + j] + cbo[q];
      float gz = g[row * 512 + 384 + j] + cbz[q];
      float i_ = expf(fminf(fmaxf(gi, -4.f), 4.f));
      float f_ = expf(fminf(fmaxf(gf, -4.f), 4.f));
      float z_ = tanhf(gz);
      float c_ = cS[row * 128 + j];
      float n_ = nS[row * 128 + j];
      c_ = fminf(fmaxf(f_ * c_ + i_ * z_, -1e6f), 1e6f);
      n_ = fminf(fmaxf(f_ * n_ + i_, 1e-6f), 1e6f);
      float h_ = (1.f / (1.f + expf(-go))) * (c_ / n_);
      if (h_ != h_) h_ = 0.f;  // nan_to_num
      cS[row * 128 + j] = c_;
      nS[row * 128 + j] = n_;
      A[row * 256 + 128 + j] = (_Float16)h_;
      xrow[j] = h_ + xr[q];
      if (last) {
        int bi = b0 + row;
        out[HS_OFF + layer * BATCH * HID + bi * HID + j] = h_;
        out[CS_OFF + layer * BATCH * HID + bi * HID + j] = c_;
        out[NS_OFF + layer * BATCH * HID + bi * HID + j] = n_;
      }
    }
    __syncthreads();
  }
}

// ---------------------------------------------------------------------------
// K3: fused self-attention, one WG per (b, head). K/V in LDS, streamed q rows.
// LDS: kS 64KB + vS 64KB + qS 1KB + aS 16KB = 145KB
// ---------------------------------------------------------------------------
__global__ __launch_bounds__(256) void attn_kernel(
    const float* __restrict__ cur, const float* __restrict__ win,
    const float* __restrict__ bin, float* __restrict__ obuf) {
  extern __shared__ char smem3[];
  float* kS = (float*)smem3;          // [512][32]
  float* vS = kS + SEQ * 32;          // [512][32]
  float* qS = vS + SEQ * 32;          // [8][32]
  float* aS = qS + 8 * 32;            // [8][512]
  const int b = blockIdx.x >> 2, head = blockIdx.x & 3;
  const int tid = threadIdx.x, wave = tid >> 5, lane = tid & 31;

  // K/V projection into LDS
  for (int idx = tid; idx < SEQ * 32; idx += 256) {
    int s = idx >> 5, d = idx & 31;
    const float* xr = cur + ((size_t)b * SEQ + s) * HID;
    int ck = HID + head * 32 + d, cv2 = 2 * HID + head * 32 + d;
    float ak = bin[ck], av = bin[cv2];
    for (int c = 0; c < HID; ++c) {
      float xv = xr[c];
      ak += xv * win[c * 3 * HID + ck];
      av += xv * win[c * 3 * HID + cv2];
    }
    kS[idx] = ak; vS[idx] = av;
  }
  __syncthreads();

  const float scale = 0.17677669529663687f;  // 1/sqrt(32)
  for (int it = 0; it < SEQ / 8; ++it) {     // uniform across waves
    int sq = it * 8 + wave;
    const float* xr = cur + ((size_t)b * SEQ + sq) * HID;
    int cq = head * 32 + lane;
    float q = bin[cq];
    for (int c = 0; c < HID; ++c) q += xr[c] * win[c * 3 * HID + cq];
    qS[wave * 32 + lane] = q;
    __syncthreads();
    float sc[16];
#pragma unroll 4
    for (int ch = 0; ch < 16; ++ch) {
      int s = ch * 32 + lane;
      float dot = 0.f;
      for (int d = 0; d < 32; ++d) dot += qS[wave * 32 + d] * kS[s * 32 + d];
      sc[ch] = dot * scale;
    }
    float mx = sc[0];
#pragma unroll
    for (int ch = 1; ch < 16; ++ch) mx = fmaxf(mx, sc[ch]);
    mx = wave_max(mx);
    float ssum = 0.f;
#pragma unroll
    for (int ch = 0; ch < 16; ++ch) { sc[ch] = expf(sc[ch] - mx); ssum += sc[ch]; }
    float inv = 1.f / wave_sum(ssum);
#pragma unroll
    for (int ch = 0; ch < 16; ++ch) aS[wave * 512 + ch * 32 + lane] = sc[ch] * inv;
    __syncthreads();
    float o = 0.f;
    for (int s = 0; s < SEQ; ++s) o += aS[wave * 512 + s] * vS[s * 32 + lane];
    obuf[((size_t)b * SEQ + sq) * HID + head * 32 + lane] = o;
  }
}

// ---------------------------------------------------------------------------
// K4: xo = LN(x + o @ attn_out_w + b) (in place over obuf) + pool logits
// ---------------------------------------------------------------------------
__global__ __launch_bounds__(256) void attnout_kernel(
    const float* __restrict__ cur, float* __restrict__ obuf,
    const float* __restrict__ wout, const float* __restrict__ bout,
    const float* __restrict__ ng, const float* __restrict__ nb,
    const float* __restrict__ pw, const float* __restrict__ pb,
    float* __restrict__ logits) {
  int row  = blockIdx.x * 8 + (threadIdx.x >> 5);
  int lane = threadIdx.x & 31;
  float*       orow = obuf + (size_t)row * HID;
  const float* xrow = cur + (size_t)row * HID;
  float val[4];
#pragma unroll
  for (int q = 0; q < 4; ++q) val[q] = bout[lane * 4 + q];
  for (int c = 0; c < HID; ++c) {
    float ov = orow[c];
    const float* wr = wout + c * HID + lane * 4;
#pragma unroll
    for (int q = 0; q < 4; ++q) val[q] += ov * wr[q];
  }
#pragma unroll
  for (int q = 0; q < 4; ++q) val[q] += xrow[lane * 4 + q];
  float m = wave_sum(val[0] + val[1] + val[2] + val[3]) * (1.0f / HID);
  float vv = 0.f;
#pragma unroll
  for (int q = 0; q < 4; ++q) { float d = val[q] - m; vv += d * d; }
  float rstd = rsqrtf(wave_sum(vv) * (1.0f / HID) + 1e-5f);
  float lgit = 0.f;
#pragma unroll
  for (int q = 0; q < 4; ++q) {
    int j = lane * 4 + q;
    float xo = (val[q] - m) * rstd * ng[j] + nb[j];
    orow[j] = xo;
    lgit += xo * pw[j];
  }
  lgit = wave_sum(lgit);
  if (lane == 0) logits[row] = lgit + pb[0];
}

// ---------------------------------------------------------------------------
// K5: softmax pooling over seq -> ctx[b, :]
// ---------------------------------------------------------------------------
__global__ __launch_bounds__(256) void pool_kernel(
    const float* __restrict__ xo, const float* __restrict__ logits,
    float* __restrict__ ctx) {
  __shared__ float wsm[SEQ];
  __shared__ float red[8];
  int b = blockIdx.x, tid = threadIdx.x;
  int wave = tid >> 5, lane = tid & 31;
  float l0 = logits[b * SEQ + tid];
  float l1 = logits[b * SEQ + 256 + tid];
  float mx = wave_max(fmaxf(l0, l1));
  if (lane == 0) red[wave] = mx;
  __syncthreads();
  float bm = red[0];
#pragma unroll
  for (int i = 1; i < 8; ++i) bm = fmaxf(bm, red[i]);
  __syncthreads();
  float e0 = expf(l0 - bm), e1 = expf(l1 - bm);
  float ssum = wave_sum(e0 + e1);
  if (lane == 0) red[wave] = ssum;
  __syncthreads();
  float Z = 0.f;
#pragma unroll
  for (int i = 0; i < 8; ++i) Z += red[i];
  float inv = 1.f / Z;
  wsm[tid] = e0 * inv;
  wsm[tid + 256] = e1 * inv;
  __syncthreads();
  if (tid < HID) {
    float acc = 0.f;
    for (int s = 0; s < SEQ; ++s)
      acc += wsm[s] * xo[((size_t)b * SEQ + s) * HID + tid];
    ctx[b * HID + tid] = acc;
  }
}

// ---------------------------------------------------------------------------
// K6: info encoder (2x Linear+LN+GELU), one 64-thread block per batch row
// ---------------------------------------------------------------------------
__global__ __launch_bounds__(64) void ie_kernel(
    const float* __restrict__ info, const float* __restrict__ w1,
    const float* __restrict__ b1, const float* __restrict__ g1,
    const float* __restrict__ b1n, const float* __restrict__ w2,
    const float* __restrict__ b2, const float* __restrict__ g2,
    const float* __restrict__ b2n, float* __restrict__ ie) {
  __shared__ float s1[64];
  __shared__ float red[2];
  int b = blockIdx.x, j = threadIdx.x;
  int wave = j >> 5, lane = j & 31;
  float t = b1[j];
  for (int c = 0; c < INFO_D; ++c) t += info[b * INFO_D + c] * w1[c * 64 + j];
  float sm = wave_sum(t);
  if (lane == 0) red[wave] = sm;
  __syncthreads();
  float m = (red[0] + red[1]) * (1.f / 64.f);
  __syncthreads();
  float d = t - m;
  float vs = wave_sum(d * d);
  if (lane == 0) red[wave] = vs;
  __syncthreads();
  float rstd = rsqrtf((red[0] + red[1]) * (1.f / 64.f) + 1e-5f);
  __syncthreads();
  s1[j] = gelu_exact(d * rstd * g1[j] + b1n[j]);
  __syncthreads();
  float t2 = b2[j];
  for (int c = 0; c < 64; ++c) t2 += s1[c] * w2[c * 64 + j];
  sm = wave_sum(t2);
  if (lane == 0) red[wave] = sm;
  __syncthreads();
  m = (red[0] + red[1]) * (1.f / 64.f);
  __syncthreads();
  d = t2 - m;
  vs = wave_sum(d * d);
  if (lane == 0) red[wave] = vs;
  __syncthreads();
  rstd = rsqrtf((red[0] + red[1]) * (1.f / 64.f) + 1e-5f);
  ie[b * 64 + j] = gelu_exact(d * rstd * g2[j] + b2n[j]);
}

// ---------------------------------------------------------------------------
// K7: backbone MLP + heads, one 256-thread block per batch row
// ---------------------------------------------------------------------------
__global__ __launch_bounds__(256) void bb_kernel(
    const float* __restrict__ ctx, const float* __restrict__ ie,
    const float* __restrict__ w1, const float* __restrict__ b1,
    const float* __restrict__ g1, const float* __restrict__ b1n,
    const float* __restrict__ w2, const float* __restrict__ b2,
    const float* __restrict__ g2, const float* __restrict__ b2n,
    const float* __restrict__ mu_w, const float* __restrict__ mu_b,
    const float* __restrict__ ls_w, const float* __restrict__ ls_b,
    float* __restrict__ out) {
  __shared__ float zin[HID + 64];
  __shared__ float z1[256];
  __shared__ float z2[128];
  __shared__ float red[8];
  int b = blockIdx.x, tid = threadIdx.x;
  int wave = tid >> 5, lane = tid & 31;
  if (tid < HID) zin[tid] = ctx[b * HID + tid];
  else if (tid < HID + 64) zin[tid] = ie[b * 64 + (tid - HID)];
  __syncthreads();
  float t = b1[tid];
  for (int c = 0; c < HID + 64; ++c) t += zin[c] * w1[c * 256 + tid];
  float sm = wave_sum(t);
  if (lane == 0) red[wave] = sm;
  __syncthreads();
  float tot = 0.f;
#pragma unroll
  for (int i = 0; i < 8; ++i) tot += red[i];
  float m = tot * (1.f / 256.f);
  __syncthreads();
  float d = t - m;
  float vs = wave_sum(d * d);
  if (lane == 0) red[wave] = vs;
  __syncthreads();
  tot = 0.f;
#pragma unroll
  for (int i = 0; i < 8; ++i) tot += red[i];
  float rstd = rsqrtf(tot * (1.f / 256.f) + 1e-5f);
  __syncthreads();
  z1[tid] = gelu_exact(d * rstd * g1[tid] + b1n[tid]);
  __syncthreads();
  float t2 = 0.f;
  if (tid < 128) {
    t2 = b2[tid];
    for (int c = 0; c < 256; ++c) t2 += z1[c] * w2[c * 128 + tid];
  }
  float contrib = (tid < 128) ? t2 : 0.f;
  sm = wave_sum(contrib);
  if (lane == 0) red[wave] = sm;
  __syncthreads();
  tot = 0.f;
#pragma unroll
  for (int i = 0; i < 8; ++i) tot += red[i];
  m = tot * (1.f / 128.f);
  __syncthreads();
  d = t2 - m;
  float c2 = (tid < 128) ? d * d : 0.f;
  vs = wave_sum(c2);
  if (lane == 0) red[wave] = vs;
  __syncthreads();
  tot = 0.f;
#pragma unroll
  for (int i = 0; i < 8; ++i) tot += red[i];
  rstd = rsqrtf(tot * (1.f / 128.f) + 1e-5f);
  __syncthreads();
  if (tid < 128) z2[tid] = gelu_exact(d * rstd * g2[tid] + b2n[tid]);
  __syncthreads();
  if (tid < ACT) {
    float mu = mu_b[tid], ls = ls_b[tid];
    for (int c = 0; c < 128; ++c) {
      mu += z2[c] * mu_w[c * ACT + tid];
      ls += z2[c] * ls_w[c * ACT + tid];
    }
    out[MU_OFF + b * ACT + tid] = mu;
    out[LS_OFF + b * ACT + tid] = fminf(fmaxf(ls, -20.f), 2.f);
  }
}

// ---------------------------------------------------------------------------
extern "C" void kernel_launch(void* const* d_in, const int* in_sizes, int n_in,
                              void* d_out, int out_size, void* d_ws,
                              size_t ws_size, hipStream_t stream) {
  (void)in_sizes; (void)n_in; (void)out_size; (void)ws_size;
  // setup_inputs() dict order, params expanded in make_params insertion order
  const float* x         = (const float*)d_in[0];   // [128,512,64]
  const float* info      = (const float*)d_in[1];   // [128,13]
  const float* in_proj_w = (const float*)d_in[2];
  const float* in_proj_b = (const float*)d_in[3];
  const float* in_norm_g = (const float*)d_in[4];
  const float* in_norm_b = (const float*)d_in[5];
  const float* l_ln_g[2] = {(const float*)d_in[6],  (const float*)d_in[10]};
  const float* l_ln_b[2] = {(const float*)d_in[7],  (const float*)d_in[11]};
  const float* l_cw[2]   = {(const float*)d_in[8],  (const float*)d_in[12]};
  const float* l_cb[2]   = {(const float*)d_in[9],  (const float*)d_in[13]};
  const float* attn_in_w  = (const float*)d_in[14];
  const float* attn_in_b  = (const float*)d_in[15];
  const float* attn_out_w = (const float*)d_in[16];
  const float* attn_out_b = (const float*)d_in[17];
  const float* attn_norm_g= (const float*)d_in[18];
  const float* attn_norm_b= (const float*)d_in[19];
  const float* pool_w = (const float*)d_in[20];
  const float* pool_b = (const float*)d_in[21];
  const float* ie_w1  = (const float*)d_in[22];
  const float* ie_b1  = (const float*)d_in[23];
  const float* ie_g1  = (const float*)d_in[24];
  const float* ie_b1n = (const float*)d_in[25];
  const float* ie_w2  = (const float*)d_in[26];
  const float* ie_b2  = (const float*)d_in[27];
  const float* ie_g2  = (const float*)d_in[28];
  const float* ie_b2n = (const float*)d_in[29];
  const float* bb_w1  = (const float*)d_in[30];
  const float* bb_b1  = (const float*)d_in[31];
  const float* bb_g1  = (const float*)d_in[32];
  const float* bb_b1n = (const float*)d_in[33];
  const float* bb_w2  = (const float*)d_in[34];
  const float* bb_b2  = (const float*)d_in[35];
  const float* bb_g2  = (const float*)d_in[36];
  const float* bb_b2n = (const float*)d_in[37];
  const float* mu_w   = (const float*)d_in[38];
  const float* mu_b   = (const float*)d_in[39];
  const float* ls_w   = (const float*)d_in[40];
  const float* ls_b   = (const float*)d_in[41];

  float* out = (float*)d_out;
  // workspace layout (floats): cur | obuf/xo | logits | ctx | ie   (~64.3MB)
  float* cur    = (float*)d_ws;
  float* obuf   = cur + (size_t)BATCH * SEQ * HID;
  float* logits = obuf + (size_t)BATCH * SEQ * HID;
  float* ctx    = logits + BATCH * SEQ;
  float* ief    = ctx + BATCH * HID;

  inproj_ln_kernel<<<BATCH * SEQ / 8, 256, 0, stream>>>(
      x, in_proj_w, in_proj_b, in_norm_g, in_norm_b, cur);

  // persistent sLSTM scan, cw LDS-resident (312KB dynamic LDS / WGP)
  const size_t slstm_lds =
      (size_t)(512 * 256 + 16 * 256) * sizeof(_Float16) +
      (size_t)(16 * 512 + 2 * 16 * 128) * sizeof(float);
  for (int l = 0; l < 2; ++l)
    slstm_kernel<<<BATCH / 16, 512, slstm_lds, stream>>>(
        cur, l_cw[l], l_cb[l], l_ln_g[l], l_ln_b[l], out, l);

  const size_t attn_lds =
      (size_t)(2 * SEQ * 32 + 8 * 32 + 8 * SEQ) * sizeof(float);
  attn_kernel<<<BATCH * 4, 256, attn_lds, stream>>>(cur, attn_in_w, attn_in_b,
                                                    obuf);
  attnout_kernel<<<BATCH * SEQ / 8, 256, 0, stream>>>(
      cur, obuf, attn_out_w, attn_out_b, attn_norm_g, attn_norm_b, pool_w,
      pool_b, logits);
  pool_kernel<<<BATCH, 256, 0, stream>>>(obuf, logits, ctx);
  ie_kernel<<<BATCH, 64, 0, stream>>>(info, ie_w1, ie_b1, ie_g1, ie_b1n, ie_w2,
                                      ie_b2, ie_g2, ie_b2n, ief);
  bb_kernel<<<BATCH, 256, 0, stream>>>(ctx, ief, bb_w1, bb_b1, bb_g1, bb_b1n,
                                       bb_w2, bb_b2, bb_g2, bb_b2n, mu_w, mu_b,
                                       ls_w, ls_b, out);
}
